// CompactionStage_28707561406717
// MI455X (gfx1250) — matrix operand order repaired and checked by reading, not compiled, not run on hardware
//
#include <hip/hip_runtime.h>
#include <cstdint>
#include <cstddef>

// Problem constants (from reference)
#define B_    16
#define P_IN  2048
#define C_IN  128
#define C2    256      // 2*C_IN
#define C_OUT 256
#define P_OUT 512
#define KNN   16
#define EPS_  1e-5f

#define INF_F (__builtin_inff())

// Fused-GEMM tiling
#define CB    4              // queries per block
#define NCOL  (CB * KNN)     // 64 activation columns per block
#define LDC   264            // f16 column stride (256 rows + 8 pad -> bank-conflict-free b128)

typedef __attribute__((ext_vector_type(16))) _Float16 v16h;
typedef __attribute__((ext_vector_type(8)))  float    v8f;

__device__ __forceinline__ v16h ld16h(const _Float16* p) {
  // two 16B loads (16B-aligned; full 32B may be only 16B-aligned in padded LDS)
  v16h r;
  ((float4*)&r)[0] = *(const float4*)(p);
  ((float4*)&r)[1] = *(const float4*)(p + 8);
  return r;
}

__device__ __forceinline__ v8f wmma_f16(v16h a, v16h b, v8f c) {
  // D = A(16x32 f16) * B(32x16 f16) + C(16x16 f32)
  return __builtin_amdgcn_wmma_f32_16x16x32_f16(false, a, false, b, (short)0, c,
                                                false, false);
}

// A-fragment layout for 16-bit 16x32 (contiguous K order, matching B):
// lanes 0-15 hold row M=lane, K in {0..15}; lanes 16-31 hold K {16..31}.
__device__ __forceinline__ int frag_off(int o, int c) {
  int kt = c >> 5, kl = c & 31;
  int mt = o >> 4, mr = o & 15;
  int half = (kl >> 4) & 1;
  int lane = mr + (half << 4);
  int pos  = kl & 15;
  return (((kt << 4) + mt) * 32 + lane) * 16 + pos;
}

// ---------------------------------------------------------------------------
// Kernel 1: weight swizzle (f32 -> f16 WMMA A-fragment layout) + BN fold
// ---------------------------------------------------------------------------
__global__ void prep_kernel(
    const float* __restrict__ w1, const float* __restrict__ w2, const float* __restrict__ wS,
    const float* __restrict__ g1, const float* __restrict__ b1, const float* __restrict__ m1, const float* __restrict__ v1,
    const float* __restrict__ g2, const float* __restrict__ b2, const float* __restrict__ m2, const float* __restrict__ v2,
    const float* __restrict__ gS, const float* __restrict__ bS, const float* __restrict__ mS, const float* __restrict__ vS,
    _Float16* __restrict__ w1f, _Float16* __restrict__ w2f, _Float16* __restrict__ wSf,
    float* __restrict__ s1, float* __restrict__ o1,
    float* __restrict__ s2, float* __restrict__ o2,
    float* __restrict__ sS, float* __restrict__ oS)
{
  int i = blockIdx.x * 256 + threadIdx.x;
  if (i < 65536) {                                   // w1 (256x256)
    int o = i >> 8, c = i & 255;
    w1f[frag_off(o, c)] = (_Float16)w1[o * C2 + c];
  } else if (i < 131072) {                           // w2 (256x256)
    int k = i - 65536; int o = k >> 8, c = k & 255;
    w2f[frag_off(o, c)] = (_Float16)w2[o * C_OUT + c];
  } else if (i < 163840) {                           // ws (256x128)
    int k = i - 131072; int o = k >> 7, c = k & 127;
    wSf[frag_off(o, c)] = (_Float16)wS[o * C_IN + c];
  } else if (i < 164608) {                           // BN fold: y = x*s + o
    int k = i - 163840; int o = k & 255; int wh = k >> 8;
    const float *g, *bb, *mm, *vv; float *sd, *od;
    if (wh == 0)      { g = g1; bb = b1; mm = m1; vv = v1; sd = s1; od = o1; }
    else if (wh == 1) { g = g2; bb = b2; mm = m2; vv = v2; sd = s2; od = o2; }
    else              { g = gS; bb = bS; mm = mS; vv = vS; sd = sS; od = oS; }
    float s = g[o] * rsqrtf(vv[o] + EPS_);
    sd[o] = s;
    od[o] = bb[o] - mm[o] * s;
  }
}

// ---------------------------------------------------------------------------
// Kernel 2: features (B,C,P) f32 -> (B,P,C) f16 so per-point gathers are
// contiguous 256B rows (16B-chunk copies into LDS in the fused kernel)
// ---------------------------------------------------------------------------
__global__ void feat_tr_kernel(const float* __restrict__ f, _Float16* __restrict__ ft) {
  int i = blockIdx.x * 256 + threadIdx.x;      // exactly B*C_IN*P_IN threads
  int p = i & (P_IN - 1);
  int c = (i >> 11) & (C_IN - 1);
  int b = i >> 18;
  ft[((size_t)(b * P_IN + p)) * C_IN + c] = (_Float16)f[((size_t)(b * C_IN + c)) * P_IN + p];
}

// ---------------------------------------------------------------------------
// Kernel 3: farthest point sampling — one block per batch, distances in VGPRs,
// wave32 shuffle argmax with jnp.argmax first-occurrence tie-break
// ---------------------------------------------------------------------------
__global__ void fps_kernel(const float* __restrict__ coords,
                           const unsigned char* __restrict__ mask,
                           int* __restrict__ cid)
{
  const int b = blockIdx.x;
  const int t = threadIdx.x;
  __shared__ float s_bv[8];
  __shared__ int   s_bi[8];
  __shared__ int   s_cur;
  const float* X = coords + (size_t)b * 2 * P_IN;
  const float* Y = X + P_IN;
  float xr[8], yr[8], dist[8];
  bool  mk[8];
#pragma unroll
  for (int j = 0; j < 8; ++j) {
    int p = t + 256 * j;
    xr[j] = X[p]; yr[j] = Y[p];
    mk[j] = mask[b * P_IN + p] != 0;
    dist[j] = mk[j] ? INF_F : -INF_F;
  }
  if (t == 0) { cid[b * P_OUT] = 0; s_cur = 0; }   // start = first valid = 0
  __syncthreads();
  int cur = 0;
  for (int it = 1; it < P_OUT; ++it) {
    float qx = X[cur], qy = Y[cur];
    float bv = -INF_F; int bi = 0x7fffffff;
#pragma unroll
    for (int j = 0; j < 8; ++j) {
      int p = t + 256 * j;
      float dx = xr[j] - qx, dy = yr[j] - qy;
      float d = dx * dx + dy * dy;
      if (mk[j]) dist[j] = fminf(dist[j], d);
      float v = dist[j];
      if (v > bv || (v == bv && p < bi)) { bv = v; bi = p; }
    }
#pragma unroll
    for (int m = 1; m < 32; m <<= 1) {
      float ov = __shfl_xor(bv, m);
      int   oi = __shfl_xor(bi, m);
      if (ov > bv || (ov == bv && oi < bi)) { bv = ov; bi = oi; }
    }
    if ((t & 31) == 0) { s_bv[t >> 5] = bv; s_bi[t >> 5] = bi; }
    __syncthreads();
    if (t == 0) {
      for (int w = 1; w < 8; ++w)
        if (s_bv[w] > bv || (s_bv[w] == bv && s_bi[w] < bi)) { bv = s_bv[w]; bi = s_bi[w]; }
      cid[b * P_OUT + it] = bi;
      s_cur = bi;
    }
    __syncthreads();
    cur = s_cur;
  }
}

// ---------------------------------------------------------------------------
// Kernel 4: KNN — one block per (b,q); 2048 distances in LDS, 16 argmin passes
// ---------------------------------------------------------------------------
__global__ void knn_kernel(const float* __restrict__ coords,
                           const unsigned char* __restrict__ mask,
                           const int* __restrict__ cid,
                           int* __restrict__ nidx)
{
  const int b = blockIdx.y, q = blockIdx.x, t = threadIdx.x;
  __shared__ float dsh[P_IN];
  __shared__ float s_bv[8];
  __shared__ int   s_bi[8];
  const float* X = coords + (size_t)b * 2 * P_IN;
  const float* Y = X + P_IN;
  const int cq = cid[b * P_OUT + q];
  const float qx = X[cq], qy = Y[cq];
#pragma unroll
  for (int j = 0; j < 8; ++j) {
    int p = t + 256 * j;
    float dx = X[p] - qx, dy = Y[p] - qy;
    dsh[p] = mask[b * P_IN + p] ? (dx * dx + dy * dy) : INF_F;
  }
  __syncthreads();
  for (int s = 0; s < KNN; ++s) {
    float bv = INF_F; int bi = 0x7fffffff;
#pragma unroll
    for (int j = 0; j < 8; ++j) {
      int p = t + 256 * j;
      float v = dsh[p];
      if (v < bv || (v == bv && p < bi)) { bv = v; bi = p; }
    }
#pragma unroll
    for (int m = 1; m < 32; m <<= 1) {
      float ov = __shfl_xor(bv, m);
      int   oi = __shfl_xor(bi, m);
      if (ov < bv || (ov == bv && oi < bi)) { bv = ov; bi = oi; }
    }
    if ((t & 31) == 0) { s_bv[t >> 5] = bv; s_bi[t >> 5] = bi; }
    __syncthreads();
    if (t == 0) {
      for (int w = 1; w < 8; ++w)
        if (s_bv[w] < bv || (s_bv[w] == bv && s_bi[w] < bi)) { bv = s_bv[w]; bi = s_bi[w]; }
      nidx[(b * P_OUT + q) * KNN + s] = bi;
      dsh[bi] = INF_F;
    }
    __syncthreads();
  }
}

// ---------------------------------------------------------------------------
// Kernel 5: fused gather + MLP1(BN,ReLU) + MLP2(BN) + masked K-maxpool
//           + shortcut(BN) + add + ReLU.  One block per (batch, 4 queries).
// Per block: GEMM 256x256x64 (x2) + 256x128x16 shortcut, all via
// v_wmma_f32_16x16x32_f16.  8 waves tiled 4(M) x 2(N), 4x2 16x16 tiles/wave.
// __launch_bounds__(256,1): single-workgroup occupancy target so the VGPR
// allocator keeps all 64 accumulator + fragment registers resident (no spill).
// ---------------------------------------------------------------------------
__global__ void __launch_bounds__(256, 1) fused_mlp(
    const _Float16* __restrict__ feat16,
    const _Float16* __restrict__ w1f,
    const _Float16* __restrict__ w2f,
    const _Float16* __restrict__ wSf,
    const float* __restrict__ s1, const float* __restrict__ o1,
    const float* __restrict__ s2, const float* __restrict__ o2,
    const float* __restrict__ sS, const float* __restrict__ oS,
    const int* __restrict__ cid, const int* __restrict__ nidx,
    const unsigned char* __restrict__ mask,
    float* __restrict__ out)
{
  extern __shared__ char smem[];
  _Float16* xh  = (_Float16*)smem;                       // 64*264*2 = 33792 B (x, then h)
  float*    sc  = (float*)(smem + 33792);                // 256*4*4  = 4096 B shortcut
  float*    vld = (float*)(smem + 33792 + 4096);         // 64*4 neighbor-valid
  int*      cpn = (int*)(smem + 33792 + 4096 + 256);     // 64*4 neighbor point id
  int*      cpc = (int*)(smem + 33792 + 4096 + 512);     // 64*4 center point id

  const int b    = blockIdx.y;
  const int qblk = blockIdx.x;            // 0..127, 4 queries each
  const int tid  = threadIdx.x;
  const int wave = tid >> 5;
  const int lane = tid & 31;
  const int lhalf = lane >> 4;
  const int l16   = lane & 15;

  // column metadata: col n -> query n/16, neighbor n%16
  if (tid < NCOL) {
    int q = qblk * CB + (tid >> 4);
    int pn = nidx[(b * P_OUT + q) * KNN + (tid & 15)];
    cpn[tid] = pn;
    cpc[tid] = cid[b * P_OUT + q];
    vld[tid] = mask[b * P_IN + pn] ? 1.0f : 0.0f;
  }
  __syncthreads();

  // Build x (256 rows x 64 cols) column-major f16: rows 0..127 center feats,
  // rows 128..255 neighbor feats.  16B chunks from the pre-transposed features.
#pragma unroll
  for (int it = 0; it < 8; ++it) {
    int flat = tid + 256 * it;            // 64 cols * 32 chunks
    int n = flat >> 5, ch = flat & 31;
    int p = (ch < 16) ? cpc[n] : cpn[n];
    float4 v = *(const float4*)(feat16 + (size_t)(b * P_IN + p) * C_IN + (size_t)(ch & 15) * 8);
    *(float4*)(xh + n * LDC + ch * 8) = v;
  }
  __syncthreads();

  const int wm = wave & 3;                // M block: rows RM..RM+63
  const int wn = wave >> 2;               // N block: cols CN..CN+31
  const int RM = wm * 64;
  const int CN = wn * 32;
  const v8f  vzero = {};
  const v16h hzero = {};

  v8f acc[4][2];
#pragma unroll
  for (int i = 0; i < 4; ++i) { acc[i][0] = vzero; acc[i][1] = vzero; }

  // GEMM1: h = w1 @ x
  for (int kt = 0; kt < 8; ++kt) {
    v16h a[4], bb[2];
#pragma unroll
    for (int i = 0; i < 4; ++i)
      a[i] = *(const v16h*)(w1f + (((kt << 4) + (wm * 4 + i)) * 32 + lane) * 16);
#pragma unroll
    for (int j = 0; j < 2; ++j)
      bb[j] = ld16h(xh + (CN + j * 16 + l16) * LDC + (kt * 32 + lhalf * 16));
#pragma unroll
    for (int i = 0; i < 4; ++i)
#pragma unroll
      for (int j = 0; j < 2; ++j)
        acc[i][j] = wmma_f16(a[i], bb[j], acc[i][j]);
  }

  // Shortcut GEMM: ws(256x128) @ cen_feats(128 x CB); B cols 0..3 = center
  // columns (n = q*16), cols 4..15 zero-padded.
  v8f sacc[2]; sacc[0] = vzero; sacc[1] = vzero;
  for (int kt = 0; kt < 4; ++kt) {
    v16h bv = hzero;
    if (l16 < CB) bv = ld16h(xh + (l16 * 16) * LDC + (kt * 32 + lhalf * 16));
#pragma unroll
    for (int i = 0; i < 2; ++i) {
      v16h a = *(const v16h*)(wSf + (((kt << 4) + (wave * 2 + i)) * 32 + lane) * 16);
      sacc[i] = wmma_f16(a, bv, sacc[i]);
    }
  }
#pragma unroll
  for (int i = 0; i < 2; ++i) {
    int mt = wave * 2 + i;
    if (l16 < CB) {
#pragma unroll
      for (int r = 0; r < 8; ++r) {
        int o = mt * 16 + r + lhalf * 8;
        sc[o * CB + l16] = sacc[i][r] * sS[o] + oS[o];
      }
    }
  }
  __syncthreads();   // all reads of x done; sc published

  // BN1 + ReLU, write h over x (same column-major layout)
#pragma unroll
  for (int i = 0; i < 4; ++i)
#pragma unroll
    for (int j = 0; j < 2; ++j) {
      int col = CN + j * 16 + l16;
#pragma unroll
      for (int r = 0; r < 8; ++r) {
        int o = RM + i * 16 + r + lhalf * 8;
        float v = acc[i][j][r] * s1[o] + o1[o];
        xh[col * LDC + o] = (_Float16)fmaxf(v, 0.0f);
      }
    }
  __syncthreads();

  // GEMM2: msg = w2 @ h
#pragma unroll
  for (int i = 0; i < 4; ++i) { acc[i][0] = vzero; acc[i][1] = vzero; }
  for (int kt = 0; kt < 8; ++kt) {
    v16h a[4], bb[2];
#pragma unroll
    for (int i = 0; i < 4; ++i)
      a[i] = *(const v16h*)(w2f + (((kt << 4) + (wm * 4 + i)) * 32 + lane) * 16);
#pragma unroll
    for (int j = 0; j < 2; ++j)
      bb[j] = ld16h(xh + (CN + j * 16 + l16) * LDC + (kt * 32 + lhalf * 16));
#pragma unroll
    for (int i = 0; i < 4; ++i)
#pragma unroll
      for (int j = 0; j < 2; ++j)
        acc[i][j] = wmma_f16(a[i], bb[j], acc[i][j]);
  }

  // BN2 + masked max over K (tile N-dim == one query's 16 neighbors):
  // D layout: lanes 0-15 row r, lanes 16-31 row r+8; N = lane%16.
#pragma unroll
  for (int i = 0; i < 4; ++i)
#pragma unroll
    for (int j = 0; j < 2; ++j) {
      int ql = wn * 2 + j;                       // local query 0..3
      float vm = vld[ql * 16 + l16];
#pragma unroll
      for (int r = 0; r < 8; ++r) {
        int o = RM + i * 16 + r + lhalf * 8;
        float v = acc[i][j][r] * s2[o] + o2[o];
        v = (vm > 0.5f) ? v : -INF_F;
        v = fmaxf(v, __shfl_xor(v, 1));
        v = fmaxf(v, __shfl_xor(v, 2));
        v = fmaxf(v, __shfl_xor(v, 4));
        v = fmaxf(v, __shfl_xor(v, 8));
        if (l16 == 0) {
          float p = (v < -3.0e38f) ? 0.0f : v;   // all-masked -> 0
          float res = fmaxf(p + sc[o * CB + ql], 0.0f);
          int qg = qblk * CB + ql;
          out[((size_t)b * C_OUT + o) * P_OUT + qg] = res;
        }
      }
    }
}

// ---------------------------------------------------------------------------
extern "C" void kernel_launch(void* const* d_in, const int* in_sizes, int n_in,
                              void* d_out, int out_size, void* d_ws, size_t ws_size,
                              hipStream_t stream)
{
  (void)in_sizes; (void)n_in; (void)out_size; (void)ws_size;
  const float* coords = (const float*)d_in[0];
  const float* feats  = (const float*)d_in[1];
  const unsigned char* mask = (const unsigned char*)d_in[2];  // jnp bool -> 1 byte
  const float* w1 = (const float*)d_in[3];
  const float* g1 = (const float*)d_in[4];
  const float* b1 = (const float*)d_in[5];
  const float* m1 = (const float*)d_in[6];
  const float* v1 = (const float*)d_in[7];
  const float* w2 = (const float*)d_in[8];
  const float* g2 = (const float*)d_in[9];
  const float* b2 = (const float*)d_in[10];
  const float* m2 = (const float*)d_in[11];
  const float* v2 = (const float*)d_in[12];
  const float* wS = (const float*)d_in[13];
  const float* gS = (const float*)d_in[14];
  const float* bS = (const float*)d_in[15];
  const float* mS = (const float*)d_in[16];
  const float* vS = (const float*)d_in[17];
  float* out = (float*)d_out;

  char* wp = (char*)d_ws;
  size_t off = 0;
  auto take = [&](size_t bytes) -> char* {
    char* p = wp + off;
    off = (off + bytes + 255) & ~(size_t)255;
    return p;
  };
  _Float16* feat16 = (_Float16*)take((size_t)B_ * P_IN * C_IN * 2);  // 8 MB
  _Float16* w1f    = (_Float16*)take(65536 * 2);
  _Float16* w2f    = (_Float16*)take(65536 * 2);
  _Float16* wSf    = (_Float16*)take(32768 * 2);
  float* s1 = (float*)take(256 * 4);
  float* o1 = (float*)take(256 * 4);
  float* s2 = (float*)take(256 * 4);
  float* o2 = (float*)take(256 * 4);
  float* sS = (float*)take(256 * 4);
  float* oS = (float*)take(256 * 4);
  int* cid  = (int*)take((size_t)B_ * P_OUT * 4);
  int* nidx = (int*)take((size_t)B_ * P_OUT * KNN * 4);

  prep_kernel<<<643, 256, 0, stream>>>(w1, w2, wS,
                                       g1, b1, m1, v1,
                                       g2, b2, m2, v2,
                                       gS, bS, mS, vS,
                                       w1f, w2f, wSf,
                                       s1, o1, s2, o2, sS, oS);
  feat_tr_kernel<<<(B_ * C_IN * P_IN) / 256, 256, 0, stream>>>(feats, feat16);
  fps_kernel<<<B_, 256, 0, stream>>>(coords, mask, cid);
  knn_kernel<<<dim3(P_OUT, B_), 256, 0, stream>>>(coords, mask, cid, nidx);

  size_t smem = 33792 + 4096 + 256 + 256 + 256;   // 38656 B dynamic LDS
  fused_mlp<<<dim3(P_OUT / CB, B_), 256, smem, stream>>>(
      feat16, w1f, w2f, wSf, s1, o1, s2, o2, sS, oS, cid, nidx, mask, out);
}
